// GeoTransolver_49435073577189
// MI455X (gfx1250) — matrix operand order repaired and checked by reading, not compiled, not run on hardware
//
#include <hip/hip_runtime.h>
#include <hip/hip_bf16.h>

// GeoTransolver forward for MI455X (gfx1250), wave32 + WMMA f32_16x16x32_f16.
// B=2, N=32768, C=256, H=8, D=32, M=32, L=4, MR=4, CIN=64, COUT=3.

typedef __attribute__((ext_vector_type(16))) _Float16 v16h;
typedef __attribute__((ext_vector_type(8)))  _Float16 v8h;
typedef __attribute__((ext_vector_type(8)))  float    v8f;

#define B_   2
#define N_   32768
#define BN_  65536
#define H_   8
#define D_   32
#define M_   32
#define L_   4
#define C_   256

__device__ __forceinline__ float gelu_tanh(float x) {
  float x3 = x * x * x;
  float t  = tanhf(0.7978845608028654f * (x + 0.044715f * x3));
  return 0.5f * x * (1.0f + t);
}

union frag16 { v16h v; v8h h[2]; };

// Load A fragment for row r (absolute), K offset kk. hi = lane>=16.
__device__ __forceinline__ v16h load_a(const _Float16* __restrict__ A, size_t row,
                                       int lda, int kk, int hi) {
  const _Float16* p = A + row * (size_t)lda + hi * 8 + kk;
  frag16 f;
  f.h[0] = *(const v8h*)(p);        // K = kk+(0|8) .. +7
  f.h[1] = *(const v8h*)(p + 16);   // K = kk+16+(0|8) .. +7
  return f.v;
}
__device__ __forceinline__ v16h load_b(const _Float16* __restrict__ Bt, size_t col,
                                       int ldb, int kk, int hi) {
  const _Float16* p = Bt + col * (size_t)ldb + hi * 8 + kk;
  frag16 f;
  f.h[0] = *(const v8h*)(p);        // K = kk+(0|8) .. +7
  f.h[1] = *(const v8h*)(p + 16);   // K = kk+16+(0|8) .. +7
  return f.v;
}

// ---------------------------------------------------------------------------
// 1x1-tile WMMA GEMM (used only for the small tok reduction, rows=32).
// ---------------------------------------------------------------------------
template <int OMODE, bool GELU>
__launch_bounds__(256)
__global__ void k_gemm1(const _Float16* __restrict__ A, int lda,
                        const _Float16* __restrict__ Bt, int ldb, int K,
                        const float* __restrict__ bias1,
                        const float* __restrict__ bias2,
                        float* __restrict__ outF, _Float16* __restrict__ outH,
                        _Float16* __restrict__ outT,
                        int ldc, long ldT, int rowTiles, int colTiles, int ncols) {
  int wid = blockIdx.x * 8 + (threadIdx.x >> 5);
  if (wid >= rowTiles * colTiles) return;
  int rowTile = wid / colTiles;
  int colTile = wid - rowTile * colTiles;
  int lane = threadIdx.x & 31;
  int hi = lane >> 4, l15 = lane & 15;

  v8f acc = {};
  for (int kk = 0; kk < K; kk += 32) {
    v16h a = load_a(A, (size_t)rowTile * 16 + l15, lda, kk, hi);
    v16h b = load_b(Bt, (size_t)colTile * 16 + l15, ldb, kk, hi);
    acc = __builtin_amdgcn_wmma_f32_16x16x32_f16(false, a, false, b,
                                                 (short)0, acc, false, false);
  }
  int nn = colTile * 16 + l15;
  bool nok = (nn < ncols);
  float bb = 0.f;
  if (bias1 && nok) bb += bias1[nn];
  if (bias2 && nok) bb += bias2[nn];
#pragma unroll
  for (int v = 0; v < 8; ++v) {
    int r = rowTile * 16 + (hi ? (8 + v) : v);
    float val = acc[v] + bb;
    if (GELU) val = gelu_tanh(val);
    if (nok) {
      size_t idx = (size_t)r * ldc + nn;
      if (OMODE == 0) outF[idx] = val;
      else if (OMODE == 1) outF[idx] += val;
      else if (OMODE == 2) outH[idx] = (_Float16)val;
    }
  }
}

// ---------------------------------------------------------------------------
// 2x2-tile (32x32 per wave) WMMA GEMM: the workhorse. rows%32==0, cols%32==0.
// OMODE: 0 = f32 store, 1 = f32 +=, 2 = f16 store, 3 = f16 transposed store
// ---------------------------------------------------------------------------
template <int OMODE, bool GELU>
__launch_bounds__(256)
__global__ void k_gemm2(const _Float16* __restrict__ A, int lda,
                        const _Float16* __restrict__ Bt, int ldb, int K,
                        const float* __restrict__ bias1,
                        const float* __restrict__ bias2,
                        float* __restrict__ outF, _Float16* __restrict__ outH,
                        _Float16* __restrict__ outT,
                        int ldc, long ldT, int rowTiles2, int colTiles2, int ncols) {
  int wid = blockIdx.x * 8 + (threadIdx.x >> 5);
  if (wid >= rowTiles2 * colTiles2) return;
  int rt = wid / colTiles2;
  int ct = wid - rt * colTiles2;
  int lane = threadIdx.x & 31;
  int hi = lane >> 4, l15 = lane & 15;
  size_t r0 = (size_t)rt * 32 + l15, r1 = r0 + 16;
  size_t c0 = (size_t)ct * 32 + l15, c1 = c0 + 16;

  v8f acc00 = {}, acc01 = {}, acc10 = {}, acc11 = {};
  for (int kk = 0; kk < K; kk += 32) {
    v16h a0 = load_a(A, r0, lda, kk, hi);
    v16h a1 = load_a(A, r1, lda, kk, hi);
    v16h b0 = load_b(Bt, c0, ldb, kk, hi);
    v16h b1 = load_b(Bt, c1, ldb, kk, hi);
    acc00 = __builtin_amdgcn_wmma_f32_16x16x32_f16(false, a0, false, b0, (short)0, acc00, false, false);
    acc01 = __builtin_amdgcn_wmma_f32_16x16x32_f16(false, a0, false, b1, (short)0, acc01, false, false);
    acc10 = __builtin_amdgcn_wmma_f32_16x16x32_f16(false, a1, false, b0, (short)0, acc10, false, false);
    acc11 = __builtin_amdgcn_wmma_f32_16x16x32_f16(false, a1, false, b1, (short)0, acc11, false, false);
  }

#pragma unroll
  for (int i = 0; i < 2; ++i) {
#pragma unroll
    for (int j = 0; j < 2; ++j) {
      const v8f& acc = (i == 0) ? (j == 0 ? acc00 : acc01) : (j == 0 ? acc10 : acc11);
      int rowbase = rt * 32 + i * 16;
      int nn = ct * 32 + j * 16 + l15;
      bool nok = (nn < ncols);
      if (OMODE == 3) {
        v8h hv;
#pragma unroll
        for (int v = 0; v < 8; ++v) hv[v] = (_Float16)acc[v];
        size_t base = (size_t)nn * ldT + (size_t)rowbase + (hi ? 8 : 0);
        *(v8h*)(outT + base) = hv;
      } else {
        float bb = 0.f;
        if (bias1 && nok) bb += bias1[nn];
        if (bias2 && nok) bb += bias2[nn];
#pragma unroll
        for (int v = 0; v < 8; ++v) {
          int r = rowbase + (hi ? (8 + v) : v);
          float val = acc[v] + bb;
          if (GELU) val = gelu_tanh(val);
          if (nok) {
            size_t idx = (size_t)r * ldc + nn;
            if (OMODE == 0) outF[idx] = val;
            else if (OMODE == 1) outF[idx] += val;
            else if (OMODE == 2) outH[idx] = (_Float16)val;
          }
        }
      }
    }
  }
}

// f32 -> f16 convert with per-row zero padding (cin -> kpad columns)
__global__ void k_cvt_pad(const float* __restrict__ in, _Float16* __restrict__ out,
                          int cin, int kpad, size_t total) {
  size_t i = (size_t)blockIdx.x * 256 + threadIdx.x;
  if (i >= total) return;
  size_t r = i / kpad;
  int c = (int)(i - r * kpad);
  out[i] = (c < cin) ? (_Float16)in[r * cin + c] : (_Float16)0.f;
}

// W (K x N, f32 row-major) -> Bt (Npad x Kpad, f16), zero padded
__global__ void k_transpose_w(const float* __restrict__ W, int K, int N,
                              _Float16* __restrict__ Bt, int Kpad, int Npad) {
  size_t total = (size_t)Npad * Kpad;
  size_t i = (size_t)blockIdx.x * 256 + threadIdx.x;
  if (i >= total) return;
  int n = (int)(i / Kpad), k = (int)(i % Kpad);
  Bt[i] = (k < K && n < N) ? (_Float16)W[(size_t)k * N + n] : (_Float16)0.f;
}

// LayerNorm over C=256, one wave per row, f16 output
__launch_bounds__(256)
__global__ void k_layernorm(const float* __restrict__ x, const float* __restrict__ g,
                            const float* __restrict__ bta, _Float16* __restrict__ out) {
  int wave = threadIdx.x >> 5, lane = threadIdx.x & 31;
  size_t row = (size_t)blockIdx.x * 8 + wave;
  const float* xr = x + row * C_;
  float v[8], s = 0.f, s2 = 0.f;
#pragma unroll
  for (int j = 0; j < 8; ++j) {
    float t = xr[lane + j * 32];
    v[j] = t; s += t; s2 += t * t;
  }
  for (int o = 16; o > 0; o >>= 1) { s += __shfl_xor(s, o, 32); s2 += __shfl_xor(s2, o, 32); }
  float mu  = s * (1.f / C_);
  float var = s2 * (1.f / C_) - mu * mu;
  float rs  = rsqrtf(var + 1e-5f);
  _Float16* orow = out + row * C_;
#pragma unroll
  for (int j = 0; j < 8; ++j) {
    int c = lane + j * 32;
    orow[c] = (_Float16)((v[j] - mu) * rs * g[c] + bta[c]);
  }
}

// Slice softmax: per (token, head) compute softmax_m(xm . Wslice / temp).
// Writes w in (h, t, m) f16 and (h, b, m, t) f16 layouts.
__launch_bounds__(256)
__global__ void k_slice_softmax(const float* __restrict__ xm, const float* __restrict__ Ws,
                                const float* __restrict__ temp,
                                _Float16* __restrict__ w_tm, _Float16* __restrict__ wT) {
  __shared__ float sWs[1024];
  for (int i = threadIdx.x; i < 1024; i += 256) sWs[i] = Ws[i];
  __syncthreads();
  int wave = threadIdx.x >> 5, lane = threadIdx.x & 31;
  size_t p = (size_t)blockIdx.x * 8 + wave;   // pair = trow*H + h
  int h = (int)(p & 7);
  size_t trow = p >> 3;
  float xv = xm[trow * C_ + h * 32 + lane];
  float lg = 0.f;
#pragma unroll
  for (int d = 0; d < 32; ++d) lg += __shfl(xv, d, 32) * sWs[d * 32 + lane];
  if (temp) lg /= temp[h];
  float mx = lg;
  for (int o = 16; o > 0; o >>= 1) mx = fmaxf(mx, __shfl_xor(mx, o, 32));
  float e = __expf(lg - mx), su = e;
  for (int o = 16; o > 0; o >>= 1) su += __shfl_xor(su, o, 32);
  float w = e / su;
  w_tm[((size_t)h * BN_ + trow) * M_ + lane] = (_Float16)w;
  size_t b = trow / N_, tl = trow - b * N_;
  wT[(((size_t)h * B_ + b) * M_ + lane) * N_ + tl] = (_Float16)w;
}

// wsum[b][h][m] = sum_t w_tm[h][b*N+t][m]   (deterministic tree reduction)
__launch_bounds__(256)
__global__ void k_wsum(const _Float16* __restrict__ w_tm, float* __restrict__ wsum) {
  __shared__ float red[256];
  int m = blockIdx.x & 31, b = (blockIdx.x >> 5) & 1, h = blockIdx.x >> 6;
  const _Float16* base = w_tm + ((size_t)h * BN_ + (size_t)b * N_) * M_ + m;
  float s = 0.f;
  for (int t = threadIdx.x; t < N_; t += 256) s += (float)base[(size_t)t * M_];
  red[threadIdx.x] = s;
  __syncthreads();
  for (int o = 128; o > 0; o >>= 1) {
    if (threadIdx.x < o) red[threadIdx.x] += red[threadIdx.x + o];
    __syncthreads();
  }
  if (threadIdx.x == 0) wsum[(b * H_ + h) * M_ + m] = red[0];
}

// tok = tok_raw / (wsum + 1e-6)
__global__ void k_tok_finalize(const float* __restrict__ tok_raw,
                               const float* __restrict__ wsum,
                               float* __restrict__ tok) {
  int i = blockIdx.x * 256 + threadIdx.x;
  if (i >= B_ * H_ * M_ * D_) return;
  tok[i] = tok_raw[i] / (wsum[i >> 5] + 1e-6f);
}

// Global-context tokens (NG=1): direct per-batch computation
__launch_bounds__(256)
__global__ void k_glob_tok(const float* __restrict__ ge, const float* __restrict__ Wx,
                           const float* __restrict__ Wfx, const float* __restrict__ Ws,
                           float* __restrict__ out) {
  __shared__ float xm[256], fx[256], sWs[1024];
  int b = blockIdx.x, tid = threadIdx.x;
  {
    float sx = 0.f, sf = 0.f;
    for (int i = 0; i < 16; ++i) {
      float g = ge[b * 16 + i];
      sx += g * Wx[i * 256 + tid];
      sf += g * Wfx[i * 256 + tid];
    }
    xm[tid] = sx; fx[tid] = sf;
  }
  for (int i = tid; i < 1024; i += 256) sWs[i] = Ws[i];
  __syncthreads();
  int h = tid >> 5, m = tid & 31;
  float lg = 0.f;
  for (int d = 0; d < 32; ++d) lg += xm[h * 32 + d] * sWs[d * 32 + m];
  float mx = lg;
  for (int o = 16; o > 0; o >>= 1) mx = fmaxf(mx, __shfl_xor(mx, o, 32));
  float e = __expf(lg - mx), su = e;
  for (int o = 16; o > 0; o >>= 1) su += __shfl_xor(su, o, 32);
  float w = e / su, inv = w / (w + 1e-6f);
  for (int d = 0; d < 32; ++d)
    out[((size_t)(b * H_ + h) * M_ + m) * D_ + d] = inv * fx[h * 32 + d];
}

// Per-(b,h) slice attention: q=tok@Wq, kv=[tok;geo;glob]@{Wk,Wv},
// softmax(q k^T / sqrt(D)) v -> otT (f16, [d][m] per bh).
__launch_bounds__(128)
__global__ void k_attn(const float* __restrict__ tok, const float* __restrict__ geo_tok,
                       const float* __restrict__ glob_tok,
                       const float* __restrict__ Wq, const float* __restrict__ Wk,
                       const float* __restrict__ Wv, _Float16* __restrict__ otT) {
  __shared__ float kv[96 * 32], qm[32 * 32], km[96 * 32], vm[96 * 32], sc[32 * 96];
  int bh = blockIdx.x, tid = threadIdx.x;
  for (int i = tid; i < 96 * 32; i += 128) {
    int grp = i >> 10;
    const float* src = (grp == 0) ? tok : (grp == 1) ? geo_tok : glob_tok;
    kv[i] = src[(size_t)bh * 1024 + (i & 1023)];
  }
  __syncthreads();
  for (int i = tid; i < 96 * 32; i += 128) {
    int r = i >> 5, d = i & 31;
    float sk = 0.f, sv = 0.f;
    for (int dd = 0; dd < 32; ++dd) {
      float t = kv[r * 32 + dd];
      sk += t * Wk[dd * 32 + d];
      sv += t * Wv[dd * 32 + d];
    }
    km[i] = sk; vm[i] = sv;
    if (r < 32) {
      float sq = 0.f;
      for (int dd = 0; dd < 32; ++dd) sq += kv[r * 32 + dd] * Wq[dd * 32 + d];
      qm[i] = sq;
    }
  }
  __syncthreads();
  const float scale = 0.17677669529663687f;  // 1/sqrt(32)
  for (int i = tid; i < 32 * 96; i += 128) {
    int m = i / 96, r = i - m * 96;
    float s = 0.f;
    for (int d = 0; d < 32; ++d) s += qm[m * 32 + d] * km[r * 32 + d];
    sc[i] = s * scale;
  }
  __syncthreads();
  if (tid < 32) {
    float mx = -1e30f;
    for (int r = 0; r < 96; ++r) mx = fmaxf(mx, sc[tid * 96 + r]);
    float su = 0.f;
    for (int r = 0; r < 96; ++r) { float e = __expf(sc[tid * 96 + r] - mx); sc[tid * 96 + r] = e; su += e; }
    float inv = 1.f / su;
    for (int r = 0; r < 96; ++r) sc[tid * 96 + r] *= inv;
  }
  __syncthreads();
  for (int i = tid; i < 32 * 32; i += 128) {
    int m = i >> 5, d = i & 31;
    float s = 0.f;
    for (int r = 0; r < 96; ++r) s += sc[m * 96 + r] * vm[r * 32 + d];
    otT[(size_t)bh * 1024 + d * 32 + m] = (_Float16)s;  // transposed: [d][m]
  }
}

// ---------------------------------------------------------------------------
// Host orchestration
// ---------------------------------------------------------------------------
template <int OMODE, bool GELU>
static inline void gemm1(hipStream_t s, const _Float16* A, int lda,
                         const _Float16* Bt, int ldb, int K,
                         const float* b1, const float* b2,
                         float* outF, _Float16* outH, _Float16* outT,
                         int ldc, long ldT, int rows, int cols, int ncols) {
  int rowTiles = rows / 16, colTiles = (cols + 15) / 16;
  long tiles = (long)rowTiles * colTiles;
  int blocks = (int)((tiles + 7) / 8);
  k_gemm1<OMODE, GELU><<<blocks, 256, 0, s>>>(A, lda, Bt, ldb, K, b1, b2,
                                              outF, outH, outT, ldc, ldT,
                                              rowTiles, colTiles, ncols);
}

template <int OMODE, bool GELU>
static inline void gemm2(hipStream_t s, const _Float16* A, int lda,
                         const _Float16* Bt, int ldb, int K,
                         const float* b1, const float* b2,
                         float* outF, _Float16* outH, _Float16* outT,
                         int ldc, long ldT, int rows, int cols, int ncols) {
  int rowTiles2 = rows / 32, colTiles2 = cols / 32;
  long tiles = (long)rowTiles2 * colTiles2;
  int blocks = (int)((tiles + 7) / 8);
  k_gemm2<OMODE, GELU><<<blocks, 256, 0, s>>>(A, lda, Bt, ldb, K, b1, b2,
                                              outF, outH, outT, ldc, ldT,
                                              rowTiles2, colTiles2, ncols);
}

static inline void cvt_pad(hipStream_t s, const float* in, _Float16* out,
                           int cin, int kpad, size_t rows) {
  size_t total = rows * kpad;
  k_cvt_pad<<<(int)((total + 255) / 256), 256, 0, s>>>(in, out, cin, kpad, total);
}

static inline void transw(hipStream_t s, const float* W, int K, int N,
                          _Float16* Bt, int Kpad, int Npad) {
  size_t total = (size_t)Npad * Kpad;
  k_transpose_w<<<(int)((total + 255) / 256), 256, 0, s>>>(W, K, N, Bt, Kpad, Npad);
}

extern "C" void kernel_launch(void* const* d_in, const int* in_sizes, int n_in,
                              void* d_out, int out_size, void* d_ws, size_t ws_size,
                              hipStream_t stream) {
  const float* local_e = (const float*)d_in[0];
  const float* geom    = (const float*)d_in[1];
  const float* glob_e  = (const float*)d_in[2];
  const float* W_in1   = (const float*)d_in[3];
  const float* b_in1   = (const float*)d_in[4];
  const float* W_in2   = (const float*)d_in[5];
  const float* b_in2   = (const float*)d_in[6];
  const float* placeh  = (const float*)d_in[7];
  const float* geo_Wx  = (const float*)d_in[8];
  const float* geo_Wfx = (const float*)d_in[9];
  const float* geo_Wsl = (const float*)d_in[10];
  const float* glob_Wx  = (const float*)d_in[11];
  const float* glob_Wfx = (const float*)d_in[12];
  const float* glob_Wsl = (const float*)d_in[13];
  const float* ln1_g = (const float*)d_in[14];
  const float* ln1_b = (const float*)d_in[15];
  const float* Wx    = (const float*)d_in[16];
  const float* Wfx   = (const float*)d_in[17];
  const float* Wsl   = (const float*)d_in[18];
  const float* temp  = (const float*)d_in[19];
  const float* Wq    = (const float*)d_in[20];
  const float* Wk    = (const float*)d_in[21];
  const float* Wv    = (const float*)d_in[22];
  const float* Wo    = (const float*)d_in[23];
  const float* ln2_g = (const float*)d_in[24];
  const float* ln2_b = (const float*)d_in[25];
  const float* Wm1   = (const float*)d_in[26];
  const float* bm1   = (const float*)d_in[27];
  const float* Wm2   = (const float*)d_in[28];
  const float* bm2   = (const float*)d_in[29];
  const float* W_o1  = (const float*)d_in[30];
  const float* b_o1  = (const float*)d_in[31];
  const float* W_o2  = (const float*)d_in[32];
  const float* b_o2  = (const float*)d_in[33];

  uint8_t* w8 = (uint8_t*)d_ws;
  size_t off = 0;
  auto alloc = [&](size_t bytes) -> void* {
    void* p = w8 + off;
    off = (off + bytes + 255) & ~(size_t)255;
    return p;
  };

  _Float16* Win1T  = (_Float16*)alloc((size_t)512 * 64 * 2);
  _Float16* Win2T  = (_Float16*)alloc((size_t)256 * 512 * 2);
  _Float16* WxT    = (_Float16*)alloc((size_t)L_ * 256 * 256 * 2);
  _Float16* WfxT   = (_Float16*)alloc((size_t)L_ * 256 * 256 * 2);
  _Float16* WoT    = (_Float16*)alloc((size_t)L_ * 256 * 256 * 2);
  _Float16* Wm1T   = (_Float16*)alloc((size_t)L_ * 1024 * 256 * 2);
  _Float16* Wm2T   = (_Float16*)alloc((size_t)L_ * 256 * 1024 * 2);
  _Float16* Wo1T   = (_Float16*)alloc((size_t)512 * 256 * 2);
  _Float16* Wo2T   = (_Float16*)alloc((size_t)32 * 512 * 2);
  _Float16* gWxT   = (_Float16*)alloc((size_t)256 * 32 * 2);
  _Float16* gWfxT  = (_Float16*)alloc((size_t)256 * 32 * 2);
  _Float16* local16 = (_Float16*)alloc((size_t)BN_ * 64 * 2);
  _Float16* geo16   = (_Float16*)alloc((size_t)BN_ * 32 * 2);
  float*    x       = (float*)alloc((size_t)BN_ * C_ * 4);
  _Float16* xh      = (_Float16*)alloc((size_t)BN_ * C_ * 2);
  _Float16* hbuf    = (_Float16*)alloc((size_t)BN_ * 1024 * 2);
  float*    xm      = (float*)alloc((size_t)BN_ * C_ * 4);
  _Float16* fxT     = (_Float16*)alloc((size_t)C_ * BN_ * 2);
  _Float16* obuf    = (_Float16*)alloc((size_t)BN_ * C_ * 2);
  _Float16* w_tm    = (_Float16*)alloc((size_t)H_ * BN_ * M_ * 2);
  _Float16* wT      = (_Float16*)alloc((size_t)H_ * BN_ * M_ * 2);
  float*    tok_raw = (float*)alloc((size_t)B_ * H_ * M_ * D_ * 4);
  float*    tok     = (float*)alloc((size_t)B_ * H_ * M_ * D_ * 4);
  float*    wsum    = (float*)alloc((size_t)B_ * H_ * M_ * 4);
  float*    geo_tok = (float*)alloc((size_t)B_ * H_ * M_ * D_ * 4);
  float*    glb_tok = (float*)alloc((size_t)B_ * H_ * M_ * D_ * 4);
  _Float16* otT     = (_Float16*)alloc((size_t)B_ * H_ * M_ * D_ * 2);

  // ---- weight conversion / transposition (f32 -> f16, B^T layout) ----
  transw(stream, W_in1, 64, 512, Win1T, 64, 512);
  transw(stream, W_in2, 512, 256, Win2T, 512, 256);
  transw(stream, W_o1, 256, 512, Wo1T, 256, 512);
  transw(stream, W_o2, 512, 3, Wo2T, 512, 32);          // pad cols 3 -> 32
  transw(stream, geo_Wx, 3, 256, gWxT, 32, 256);        // pad K 3 -> 32
  transw(stream, geo_Wfx, 3, 256, gWfxT, 32, 256);
  for (int l = 0; l < L_; ++l) {
    transw(stream, Wx + (size_t)l * 256 * 256, 256, 256, WxT + (size_t)l * 256 * 256, 256, 256);
    transw(stream, Wfx + (size_t)l * 256 * 256, 256, 256, WfxT + (size_t)l * 256 * 256, 256, 256);
    transw(stream, Wo + (size_t)l * 256 * 256, 256, 256, WoT + (size_t)l * 256 * 256, 256, 256);
    transw(stream, Wm1 + (size_t)l * 256 * 1024, 256, 1024, Wm1T + (size_t)l * 1024 * 256, 256, 1024);
    transw(stream, Wm2 + (size_t)l * 1024 * 256, 1024, 256, Wm2T + (size_t)l * 256 * 1024, 1024, 256);
  }
  cvt_pad(stream, local_e, local16, 64, 64, BN_);
  cvt_pad(stream, geom, geo16, 3, 32, BN_);

  // ---- preprocess MLP: x = gelu(local@W_in1+b1)@W_in2 + b2 + placeholder ----
  gemm2<2, true>(stream, local16, 64, Win1T, 64, 64, b_in1, nullptr,
                 nullptr, hbuf, nullptr, 512, 0, BN_, 512, 512);
  gemm2<0, false>(stream, hbuf, 512, Win2T, 512, 512, b_in2, placeh,
                  x, nullptr, nullptr, 256, 0, BN_, 256, 256);

  // ---- geometry context tokens ----
  gemm2<0, false>(stream, geo16, 32, gWxT, 32, 32, nullptr, nullptr,
                  xm, nullptr, nullptr, 256, 0, BN_, 256, 256);
  gemm2<3, false>(stream, geo16, 32, gWfxT, 32, 32, nullptr, nullptr,
                  nullptr, nullptr, fxT, 0, BN_, BN_, 256, 256);
  k_slice_softmax<<<BN_ * H_ / 8, 256, 0, stream>>>(xm, geo_Wsl, nullptr, w_tm, wT);
  for (int b = 0; b < B_; ++b)
    for (int h = 0; h < H_; ++h)
      gemm1<0, false>(stream, wT + (((size_t)h * B_ + b) * M_) * N_, N_,
                      fxT + (size_t)(h * 32) * BN_ + (size_t)b * N_, BN_, N_,
                      nullptr, nullptr, tok_raw + (size_t)(b * H_ + h) * 1024,
                      nullptr, nullptr, 32, 0, 32, 32, 32);
  k_wsum<<<B_ * H_ * M_, 256, 0, stream>>>(w_tm, wsum);
  k_tok_finalize<<<64, 256, 0, stream>>>(tok_raw, wsum, geo_tok);

  // ---- global context tokens (NG = 1) ----
  k_glob_tok<<<B_, 256, 0, stream>>>(glob_e, glob_Wx, glob_Wfx, glob_Wsl, glb_tok);

  // ---- transformer layers ----
  for (int l = 0; l < L_; ++l) {
    k_layernorm<<<BN_ / 8, 256, 0, stream>>>(x, ln1_g + l * 256, ln1_b + l * 256, xh);
    gemm2<0, false>(stream, xh, 256, WxT + (size_t)l * 256 * 256, 256, 256,
                    nullptr, nullptr, xm, nullptr, nullptr, 256, 0, BN_, 256, 256);
    gemm2<3, false>(stream, xh, 256, WfxT + (size_t)l * 256 * 256, 256, 256,
                    nullptr, nullptr, nullptr, nullptr, fxT, 0, BN_, BN_, 256, 256);
    k_slice_softmax<<<BN_ * H_ / 8, 256, 0, stream>>>(xm, Wsl + (size_t)l * 1024,
                                                      temp + l * H_, w_tm, wT);
    for (int b = 0; b < B_; ++b)
      for (int h = 0; h < H_; ++h)
        gemm1<0, false>(stream, wT + (((size_t)h * B_ + b) * M_) * N_, N_,
                        fxT + (size_t)(h * 32) * BN_ + (size_t)b * N_, BN_, N_,
                        nullptr, nullptr, tok_raw + (size_t)(b * H_ + h) * 1024,
                        nullptr, nullptr, 32, 0, 32, 32, 32);
    k_wsum<<<B_ * H_ * M_, 256, 0, stream>>>(w_tm, wsum);
    k_tok_finalize<<<64, 256, 0, stream>>>(tok_raw, wsum, tok);
    k_attn<<<B_ * H_, 128, 0, stream>>>(tok, geo_tok, glb_tok,
                                        Wq + (size_t)l * 1024, Wk + (size_t)l * 1024,
                                        Wv + (size_t)l * 1024, otT);
    // deslice: o[t, h*32+d] = sum_m w[t,m] * ot[m,d]  (per b,h; K = 32)
    for (int b = 0; b < B_; ++b)
      for (int h = 0; h < H_; ++h)
        gemm2<2, false>(stream, w_tm + ((size_t)h * BN_ + (size_t)b * N_) * M_, M_,
                        otT + (size_t)(b * H_ + h) * 1024, 32, 32,
                        nullptr, nullptr, nullptr,
                        obuf + (size_t)b * N_ * 256 + h * 32, nullptr,
                        256, 0, N_, 32, 32);
    // x += o @ Wo
    gemm2<1, false>(stream, obuf, 256, WoT + (size_t)l * 256 * 256, 256, 256,
                    nullptr, nullptr, x, nullptr, nullptr, 256, 0, BN_, 256, 256);
    // MLP
    k_layernorm<<<BN_ / 8, 256, 0, stream>>>(x, ln2_g + l * 256, ln2_b + l * 256, xh);
    gemm2<2, true>(stream, xh, 256, Wm1T + (size_t)l * 1024 * 256, 256, 256,
                   bm1 + (size_t)l * 1024, nullptr, nullptr, hbuf, nullptr,
                   1024, 0, BN_, 1024, 1024);
    gemm2<1, false>(stream, hbuf, 1024, Wm2T + (size_t)l * 256 * 1024, 1024, 1024,
                    bm2 + (size_t)l * 256, nullptr, x, nullptr, nullptr,
                    256, 0, BN_, 256, 256);
  }

  // ---- output head ----
  cvt_pad(stream, x, xh, 256, 256, BN_);
  gemm2<2, true>(stream, xh, 256, Wo1T, 256, 256, b_o1, nullptr,
                 nullptr, hbuf, nullptr, 512, 0, BN_, 512, 512);
  gemm2<0, false>(stream, hbuf, 512, Wo2T, 512, 512, b_o2, nullptr,
                  (float*)d_out, nullptr, nullptr, 3, 0, BN_, 32, 3);
}
